// token_merging_39633958208220
// MI455X (gfx1250) — matrix operand order repaired and checked by reading, not compiled, not run on hardware
//
#include <hip/hip_runtime.h>
#include <hip/hip_bf16.h>
#include <math.h>

typedef __attribute__((ext_vector_type(16))) _Float16 v16h;
typedef __attribute__((ext_vector_type(8)))  float    v8f;

#define EPSF 1e-8f

// ---------------------------------------------------------------------------
// WMMA helpers: CDNA5 v_wmma_f32_16x16x32_f16, wave32, D = A(16x32) * B(32x16) + C
// ---------------------------------------------------------------------------
__device__ __forceinline__ v8f wmma_f16(v16h a, v16h b, v8f c) {
  return __builtin_amdgcn_wmma_f32_16x16x32_f16(
      /*neg_a=*/false, a, /*neg_b=*/false, b,
      /*c_mod=*/(short)0, c, /*reuse_a=*/false, /*reuse_b=*/false);
}

// A-matrix gather (16-bit A 16x32, ISA 7.12.2):
// lane L (L<16): row M=L,    K = {0..7, 16..23}
// lane L (L>=16): row M=L-16, K = {8..15, 24..31}
// LDS tile: row-major [16 rows][stride_h halves]; stride_h*2 must be 16B-aligned.
__device__ __forceinline__ v16h load_A16(const _Float16* tile, int stride_h) {
  int L   = threadIdx.x & 31;
  int row = L & 15;
  int hs  = (L >> 4) & 1;
  const uint4* p = (const uint4*)(tile + row * stride_h + hs * 8); // 8 halves = 16B
  union { uint4 q[2]; v16h h; } t;
  t.q[0] = p[0];   // K chunk 0
  t.q[1] = p[2];   // +16 halves -> K chunk 1
  return t.h;
}

// B-matrix gather (16-bit B 32x16):
// lane L holds column N = L&15; K = {0..7, 16..23} (L<16) or {8..15, 24..31} (L>=16), 2 per VGPR.
// LDS tile stored with row = B-column index (contiguous K per row).
__device__ __forceinline__ v16h load_B16(const _Float16* tile, int stride_h) {
  int L   = threadIdx.x & 31;
  int row = L & 15;
  int hs  = (L >> 4) & 1;
  const uint4* p = (const uint4*)(tile + row * stride_h + hs * 8);
  union { uint4 q[2]; v16h h; } t;
  t.q[0] = p[0];
  t.q[1] = p[2];
  return t.h;
}

// ---------------------------------------------------------------------------
// Kernel 1: normalize centers (reference double-normalization chain) -> f16
// ---------------------------------------------------------------------------
__global__ __launch_bounds__(256) void tm_norm_centers(const float* __restrict__ centers,
                                                       _Float16* __restrict__ cn_h) {
  const int row = blockIdx.x;               // 0 .. B*K-1
  const float* src = centers + (size_t)row * 768;
  const int t = threadIdx.x;
  float v0 = src[t], v1 = src[t + 256], v2 = src[t + 512];
  float s = v0 * v0 + v1 * v1 + v2 * v2;
  __shared__ float red[256];
  red[t] = s; __syncthreads();
  for (int off = 128; off > 0; off >>= 1) {
    if (t < off) red[t] += red[t + off];
    __syncthreads();
  }
  float nrm = sqrtf(red[0]);
  float t1  = 1.0f / (nrm + EPSF);          // tn = c/(||c||+eps)
  float tnn = nrm * t1;                     // ||tn||
  float f   = t1 / fmaxf(tnn, EPSF);        // tn2 = tn / max(||tn||,eps)
  _Float16* dst = cn_h + (size_t)row * 768;
  dst[t]       = (_Float16)(v0 * f);
  dst[t + 256] = (_Float16)(v1 * f);
  dst[t + 512] = (_Float16)(v2 * f);
}

// ---------------------------------------------------------------------------
// Kernel 2: per-token combined normalization factor -> f32
// ---------------------------------------------------------------------------
__global__ __launch_bounds__(256) void tm_token_norms(const float* __restrict__ x,
                                                      float* __restrict__ inv_tok) {
  const int row = blockIdx.x;               // 0 .. B*N-1
  const float* src = x + (size_t)row * 768;
  const int t = threadIdx.x;
  float v0 = src[t], v1 = src[t + 256], v2 = src[t + 512];
  float s = v0 * v0 + v1 * v1 + v2 * v2;
  __shared__ float red[256];
  red[t] = s; __syncthreads();
  for (int off = 128; off > 0; off >>= 1) {
    if (t < off) red[t] += red[t + off];
    __syncthreads();
  }
  if (t == 0) {
    float nrm = sqrtf(red[0]);
    float t1  = 1.0f / (nrm + EPSF);
    float tnn = nrm * t1;
    inv_tok[row] = t1 / fmaxf(tnn, EPSF);
  }
}

__global__ void tm_zero_den(float* __restrict__ den) {
  den[blockIdx.x * 256 + threadIdx.x] = 0.0f;
}

// ---------------------------------------------------------------------------
// Kernel 4: fused cosine-sim GEMM + softmax over K, writes prob^T (B,K,N) f16
// and accumulates den[b,k] = sum_n prob. Block: 256 thr (8 waves), 128 tokens.
// ---------------------------------------------------------------------------
__global__ __launch_bounds__(256) void tm_cos_softmax(const float* __restrict__ x,
                                                      const _Float16* __restrict__ cn_h,
                                                      const float* __restrict__ inv_tok,
                                                      _Float16* __restrict__ prob_t,
                                                      float* __restrict__ den) {
  constexpr int N = 8192, D = 768, K = 256;
  constexpr int MT = 128;                // tokens per workgroup
  constexpr int XS_STR = 40;             // 80B row stride: 16B aligned, conflict-free
  constexpr int CS_STR = 40;
  constexpr int PS_STR = 136;            // 272B: 16B aligned

  constexpr int XS_BYTES  = MT * XS_STR * 2;          // 10240
  constexpr int CS_BYTES  = K * CS_STR * 2;           // 20480
  constexpr int INV_BYTES = MT * 4;                   // 512
  constexpr int PROB_BYTES = K * PS_STR * 2;          // 69632
  constexpr int DEN_BYTES  = K * 4;                   // 1024
  constexpr int LOOP_BYTES = XS_BYTES + CS_BYTES + INV_BYTES;
  constexpr int SMEM_BYTES = (PROB_BYTES + DEN_BYTES) > LOOP_BYTES
                                 ? (PROB_BYTES + DEN_BYTES) : LOOP_BYTES;
  __shared__ __align__(16) unsigned char smem[SMEM_BYTES];
  _Float16* xs    = (_Float16*)smem;                               // phase 1
  _Float16* cs    = (_Float16*)(smem + XS_BYTES);                  // phase 1
  float*    inv_s = (float*)(smem + XS_BYTES + CS_BYTES);          // phase 1
  _Float16* probs = (_Float16*)smem;                               // phase 2
  float*    den_s = (float*)(smem + PROB_BYTES);                   // phase 2 (no overlap w/ phase1)

  const int b  = blockIdx.y;
  const int n0 = blockIdx.x * MT;
  const int tid = threadIdx.x;
  const int w   = tid >> 5;
  const int L   = tid & 31;
  const int ln  = L & 15;
  const int half = (L >> 4) & 1;

  den_s[tid] = 0.0f;
  if (tid < MT) inv_s[tid] = inv_tok[(size_t)b * N + n0 + tid];

  v8f acc[16] = {};   // wave w owns token rows w*16..w*16+15, all 256 centers

  for (int ch = 0; ch < D / 32; ++ch) {
    const int d0 = ch * 32;
    __syncthreads();
    // stage x tile: 128 tokens x 32 dims, f32 -> normalized f16
    {
      int r  = tid >> 1;
      int hf = tid & 1;
      const float* src = x + ((size_t)b * N + n0 + r) * D + d0 + hf * 16;
      float inv = inv_s[r];
      float4 a0 = ((const float4*)src)[0];
      float4 a1 = ((const float4*)src)[1];
      float4 a2 = ((const float4*)src)[2];
      float4 a3 = ((const float4*)src)[3];
      union { _Float16 h[16]; uint4 q[2]; } pk;
      pk.h[0]=(_Float16)(a0.x*inv); pk.h[1]=(_Float16)(a0.y*inv);
      pk.h[2]=(_Float16)(a0.z*inv); pk.h[3]=(_Float16)(a0.w*inv);
      pk.h[4]=(_Float16)(a1.x*inv); pk.h[5]=(_Float16)(a1.y*inv);
      pk.h[6]=(_Float16)(a1.z*inv); pk.h[7]=(_Float16)(a1.w*inv);
      pk.h[8]=(_Float16)(a2.x*inv); pk.h[9]=(_Float16)(a2.y*inv);
      pk.h[10]=(_Float16)(a2.z*inv); pk.h[11]=(_Float16)(a2.w*inv);
      pk.h[12]=(_Float16)(a3.x*inv); pk.h[13]=(_Float16)(a3.y*inv);
      pk.h[14]=(_Float16)(a3.z*inv); pk.h[15]=(_Float16)(a3.w*inv);
      uint4* dst = (uint4*)(xs + r * XS_STR + hf * 16);
      dst[0] = pk.q[0]; dst[1] = pk.q[1];
    }
    // stage centers tile: 256 centers x 16 dims (already normalized f16)
    {
      const _Float16* src = cn_h + ((size_t)b * K + tid) * D + d0;
      uint4 w0 = ((const uint4*)src)[0];
      uint4 w1 = ((const uint4*)src)[1];
      uint4* dst = (uint4*)(cs + tid * CS_STR);
      dst[0] = w0; dst[1] = w1;
    }
    __syncthreads();
    v16h aop = load_A16(xs + (w * 16) * XS_STR, XS_STR);
#pragma unroll
    for (int kt = 0; kt < 16; ++kt) {
      v16h bop = load_B16(cs + (kt * 16) * CS_STR, CS_STR);
      acc[kt] = wmma_f16(aop, bop, acc[kt]);
    }
  }
  __syncthreads();   // last reads of xs/cs done; smem becomes probs/den_s

  // softmax over K=256 per token row. C layout: lanes 0-15 -> M=v, lanes 16-31 -> M=v+8;
  // column = kt*16 + (L&15). Rows live within a 16-lane half -> shfl_xor masks 8..1.
#pragma unroll
  for (int v = 0; v < 8; ++v) {
    float mx = acc[0][v];
#pragma unroll
    for (int kt = 1; kt < 16; ++kt) mx = fmaxf(mx, acc[kt][v]);
    for (int off = 8; off >= 1; off >>= 1) mx = fmaxf(mx, __shfl_xor(mx, off, 32));
    float sum = 0.0f;
#pragma unroll
    for (int kt = 0; kt < 16; ++kt) {
      float e = __expf(acc[kt][v] - mx);
      acc[kt][v] = e;
      sum += e;
    }
    for (int off = 8; off >= 1; off >>= 1) sum += __shfl_xor(sum, off, 32);
    float invs = 1.0f / sum;
#pragma unroll
    for (int kt = 0; kt < 16; ++kt) acc[kt][v] *= invs;
  }

  // den partial sums + transposed prob staging
#pragma unroll
  for (int kt = 0; kt < 16; ++kt) {
    float s = 0.0f;
#pragma unroll
    for (int v = 0; v < 8; ++v) {
      s += acc[kt][v];
      probs[(kt * 16 + ln) * PS_STR + w * 16 + v + half * 8] = (_Float16)acc[kt][v];
    }
    atomicAdd(&den_s[kt * 16 + ln], s);
  }
  __syncthreads();

  // coalesced write of prob^T[b][k][n0..n0+128]
  {
    const uint4* srcr = (const uint4*)(probs + tid * PS_STR);
    uint4* dst = (uint4*)(prob_t + ((size_t)b * K + tid) * N + n0);
#pragma unroll
    for (int j = 0; j < 16; ++j) dst[j] = srcr[j];
  }
  atomicAdd(den + (size_t)b * K + tid, den_s[tid]);
}

// ---------------------------------------------------------------------------
// Kernel 5: out[b,k,d] = (sum_n prob[n,k] * x[n,d]) / (den[b,k]+eps)
// grid (D/128, K/16, B); block 256 (8 waves, each a 16x16 d-subtile).
// Stages 128 n per iteration -> 4 WMMAs per wave per barrier pair.
// ---------------------------------------------------------------------------
__global__ __launch_bounds__(256) void tm_aggregate(const _Float16* __restrict__ prob_t,
                                                    const float* __restrict__ x,
                                                    const float* __restrict__ den,
                                                    float* __restrict__ out) {
  constexpr int N = 8192, D = 768, K = 256;
  constexpr int NB = 128;                     // n-depth per stage
  constexpr int STR = 136;                    // 272B row stride: 16B aligned, conflict-free
  __shared__ __align__(16) _Float16 as[16 * STR];    // prob^T tile: 16 k x 128 n
  __shared__ __align__(16) _Float16 xst[128 * STR];  // x tile transposed: 128 d x 128 n
  __shared__ float dn[16];

  const int b  = blockIdx.z;
  const int k0 = blockIdx.y * 16;
  const int d0 = blockIdx.x * 128;
  const int tid = threadIdx.x;
  const int w   = tid >> 5;
  const int L   = tid & 31;
  const int ln  = L & 15;
  const int half = (L >> 4) & 1;

  if (tid < 16) dn[tid] = den[(size_t)b * K + k0 + tid];

  v8f acc = {};
  for (int nit = 0; nit < N / NB; ++nit) {    // 64 iterations
    const int n0 = nit * NB;
    __syncthreads();
    {   // stage A: 16 k rows x 128 n halves of prob^T (already transposed)
      int r = tid >> 4, c = (tid & 15) * 8;
      *(uint4*)(as + r * STR + c) =
          *(const uint4*)(prob_t + ((size_t)b * K + k0 + r) * N + n0 + c);
    }
    {   // stage x: 128 n x 128 d, transposed into xst[d][n], f32 -> f16
      int r  = tid >> 1;            // local n row
      int hf = tid & 1;             // d half (64 each)
      const float* src = x + ((size_t)b * N + n0 + r) * D + d0 + hf * 64;
#pragma unroll
      for (int q = 0; q < 16; ++q) {
        float4 f = ((const float4*)src)[q];
        int dl = hf * 64 + q * 4;
        xst[(dl + 0) * STR + r] = (_Float16)f.x;
        xst[(dl + 1) * STR + r] = (_Float16)f.y;
        xst[(dl + 2) * STR + r] = (_Float16)f.z;
        xst[(dl + 3) * STR + r] = (_Float16)f.w;
      }
    }
    __syncthreads();
#pragma unroll
    for (int c = 0; c < 4; ++c) {             // 4 K-chunks of 32 n
      v16h aop = load_A16(as + c * 32, STR);
      v16h bop = load_B16(xst + (w * 16) * STR + c * 32, STR);
      acc = wmma_f16(aop, bop, acc);
    }
  }

#pragma unroll
  for (int v = 0; v < 8; ++v) {
    int kl = v + half * 8;
    float o = acc[v] / (dn[kl] + EPSF);
    out[((size_t)b * K + k0 + kl) * D + d0 + w * 16 + ln] = o;
  }
}

// ---------------------------------------------------------------------------
extern "C" void kernel_launch(void* const* d_in, const int* in_sizes, int n_in,
                              void* d_out, int out_size, void* d_ws, size_t ws_size,
                              hipStream_t stream) {
  (void)in_sizes; (void)n_in; (void)out_size; (void)ws_size;
  const float* x       = (const float*)d_in[0];   // (8, 8192, 768) f32
  const float* centers = (const float*)d_in[1];   // (8, 256, 768) f32
  float* out = (float*)d_out;                     // (8, 256, 768) f32

  constexpr int B = 8, N = 8192, D = 768, K = 256;
  unsigned char* ws = (unsigned char*)d_ws;
  size_t off = 0;
  _Float16* prob_t = (_Float16*)(ws + off); off += (size_t)B * K * N * 2;   // 33.5 MB
  _Float16* cn_h   = (_Float16*)(ws + off); off += (size_t)B * K * D * 2;   //  3.1 MB
  float*    inv_tok= (float*)(ws + off);    off += (size_t)B * N * 4;       //  0.25 MB
  float*    den    = (float*)(ws + off);    off += (size_t)B * K * 4;       //  8 KB

  tm_norm_centers<<<dim3(B * K), 256, 0, stream>>>(centers, cn_h);
  tm_token_norms<<<dim3(B * N), 256, 0, stream>>>(x, inv_tok);
  tm_zero_den<<<dim3(B), 256, 0, stream>>>(den);
  tm_cos_softmax<<<dim3(N / 128, B), 256, 0, stream>>>(x, cn_h, inv_tok, prob_t, den);
  tm_aggregate<<<dim3(D / 128, K / 16, B), 256, 0, stream>>>(prob_t, x, den, out);
}
